// Qwen3MoeSparseMoeBlock_57269093925207
// MI455X (gfx1250) — matrix operand order repaired and checked by reading, not compiled, not run on hardware
//
#include <hip/hip_runtime.h>

// ---------------------------------------------------------------------------
// Qwen3 MoE sparse block for gfx1250 (MI455X), wave32 + WMMA bf16.
//   B=2, S=2048 -> T=4096 tokens, D=2048, E=16 experts, F=768, top-2 routing.
// Pipeline: zero -> router(top2) -> grouped gate/up GEMM (+SwiGLU) -> down GEMM.
// Workspace layout (needs ~13.2 MB):
//   [0,64)                      : int counts[16]
//   [256, 256+16*4096*4)        : int  expert_tokens[16][4096]
//   [.., +16*4096*4)            : f32  expert_weights[16][4096]
//   [524544, +8192*768*2)       : bf16 h buffer (compacted token-expert slots)
// ---------------------------------------------------------------------------

#define TOK  4096
#define DIM  2048
#define NEXP 16
#define NF   768

typedef __attribute__((ext_vector_type(16))) __bf16 v16bf;
typedef __attribute__((ext_vector_type(8)))  float  v8f;

__device__ __forceinline__ int prefix_count(const int* __restrict__ counts, int e) {
    int s = 0;
    for (int i = 0; i < e; ++i) s += counts[i];
    return s;
}

// Load 16 consecutive fp32 and convert to a bf16 B-fragment (K=+0..15 contiguous).
__device__ __forceinline__ v16bf load_cvt16(const float* __restrict__ p) {
    float t[16];
    *(float4*)&t[0]  = *(const float4*)(p + 0);
    *(float4*)&t[4]  = *(const float4*)(p + 4);
    *(float4*)&t[8]  = *(const float4*)(p + 8);
    *(float4*)&t[12] = *(const float4*)(p + 12);
    v16bf r;
#pragma unroll
    for (int i = 0; i < 16; ++i) r[i] = (__bf16)t[i];
    return r;
}

// ---------------------------------------------------------------------------
__global__ __launch_bounds__(256) void moe_zero(float* __restrict__ out,
                                                int* __restrict__ counts) {
    int idx = blockIdx.x * blockDim.x + threadIdx.x;
    if (idx < (TOK * DIM) / 4) ((float4*)out)[idx] = make_float4(0.f, 0.f, 0.f, 0.f);
    if (blockIdx.x == 0 && threadIdx.x < NEXP) counts[threadIdx.x] = 0;
}

// ---------------------------------------------------------------------------
// Router: 16 tokens x 16 experts per block (thread = one (token,expert) dot).
__global__ __launch_bounds__(256) void moe_router(const float* __restrict__ x,
                                                  const float* __restrict__ rw,
                                                  float* __restrict__ logits_out,
                                                  int* __restrict__ counts,
                                                  int* __restrict__ etok,
                                                  float* __restrict__ ewt) {
    __shared__ float lg[16 * 16];
    const int tid = threadIdx.x;
    const int tl  = tid >> 4;   // token within block
    const int e   = tid & 15;   // expert
    const int tok = blockIdx.x * 16 + tl;

    const float* xr = x  + (size_t)tok * DIM;
    const float* wr = rw + (size_t)e * DIM;
    float acc = 0.f;
    for (int k = 0; k < DIM; k += 4) {
        float4 a = *(const float4*)(xr + k);
        float4 b = *(const float4*)(wr + k);
        acc = fmaf(a.x, b.x, acc);
        acc = fmaf(a.y, b.y, acc);
        acc = fmaf(a.z, b.z, acc);
        acc = fmaf(a.w, b.w, acc);
    }
    logits_out[(size_t)tok * NEXP + e] = acc;
    lg[tl * 16 + e] = acc;
    __syncthreads();

    if (e == 0) {
        const float* l = &lg[tl * 16];
        float mx = l[0];
        for (int i = 1; i < NEXP; ++i) mx = fmaxf(mx, l[i]);
        float p[NEXP];
        float sum = 0.f;
        for (int i = 0; i < NEXP; ++i) { p[i] = __expf(l[i] - mx); sum += p[i]; }
        float inv = 1.f / sum;
        for (int i = 0; i < NEXP; ++i) p[i] *= inv;
        // top-2, first-index wins ties (matches jax.lax.top_k)
        int i1 = 0;
        for (int i = 1; i < NEXP; ++i) if (p[i] > p[i1]) i1 = i;
        int i2 = (i1 == 0) ? 1 : 0;
        for (int i = 0; i < NEXP; ++i) if (i != i1 && p[i] > p[i2]) i2 = i;
        float denom = p[i1] + p[i2];
        float w1 = p[i1] / denom, w2 = p[i2] / denom;
        int pos1 = atomicAdd(&counts[i1], 1);
        etok[i1 * TOK + pos1] = tok; ewt[i1 * TOK + pos1] = w1;
        int pos2 = atomicAdd(&counts[i2], 1);
        etok[i2 * TOK + pos2] = tok; ewt[i2 * TOK + pos2] = w2;
    }
}

// ---------------------------------------------------------------------------
// Grouped gate/up GEMM + SwiGLU. Block = 256 thr = 8 waves.
// Tile: M=64 tokens (expert list) x N=128 F-cols; wave w owns N strip w*16..+15.
// K-loop over D in steps of 32 using v_wmma_f32_16x16x32_bf16.
__global__ __launch_bounds__(256) void moe_gateup(const float* __restrict__ x,
                                                  const float* __restrict__ gateW,
                                                  const float* __restrict__ upW,
                                                  const int* __restrict__ etok,
                                                  const float* __restrict__ ewt,
                                                  const int* __restrict__ counts,
                                                  __bf16* __restrict__ hbuf) {
    const int e     = blockIdx.z;
    const int count = counts[e];
    const int tileM = blockIdx.y * 64;
    if (tileM >= count) return;
    const int fBase    = blockIdx.x * 128;
    const int slotBase = prefix_count(counts, e);

    __shared__ __bf16 a_lds[64 * 32];

    const int tid  = threadIdx.x;
    const int wave = tid >> 5;
    const int lane = tid & 31;

    // A-staging coords: thread t loads 8 fp32 of row sm, quarter sq
    const int sm = tid >> 2;
    const int sq = tid & 3;
    const int srow   = tileM + sm;
    const bool svalid = srow < count;
    const int stok = svalid ? etok[e * TOK + srow] : 0;
    const float* srcRow = x + (size_t)stok * DIM + sq * 8;

    // WMMA fragment coords (K contiguous per half-wave, same order in A and B)
    const int n     = lane & 15;
    const int khalf = (lane < 16) ? 0 : 16;  // B: lanes 0-15 K=0..15, 16-31 K=16..31
    const int kbase = (lane < 16) ? 0 : 16;  // A: K offset for upper half-wave
    const int arow  = (lane < 16) ? lane : (lane - 16);
    const int f     = fBase + wave * 16 + n;
    const float* gRow = gateW + ((size_t)e * NF + f) * DIM + khalf;
    const float* uRow = upW   + ((size_t)e * NF + f) * DIM + khalf;

    v8f accG[4], accU[4];
    v8f vz = {};
#pragma unroll
    for (int i = 0; i < 4; ++i) { accG[i] = vz; accU[i] = vz; }

    for (int k0 = 0; k0 < DIM; k0 += 32) {
        // stage A tile: 64 tokens x 32 K, fp32 -> bf16
        __bf16* dst = &a_lds[sm * 32 + sq * 8];
        if (svalid) {
            float4 p0 = *(const float4*)(srcRow + k0);
            float4 p1 = *(const float4*)(srcRow + k0 + 4);
            dst[0] = (__bf16)p0.x; dst[1] = (__bf16)p0.y;
            dst[2] = (__bf16)p0.z; dst[3] = (__bf16)p0.w;
            dst[4] = (__bf16)p1.x; dst[5] = (__bf16)p1.y;
            dst[6] = (__bf16)p1.z; dst[7] = (__bf16)p1.w;
        } else {
#pragma unroll
            for (int i = 0; i < 8; ++i) dst[i] = (__bf16)0.f;
        }
        __syncthreads();

        v16bf bG = load_cvt16(gRow + k0);
        v16bf bU = load_cvt16(uRow + k0);

#pragma unroll
        for (int ms = 0; ms < 4; ++ms) {
            const __bf16* ap = &a_lds[(ms * 16 + arow) * 32 + kbase];
            v16bf a;
#pragma unroll
            for (int i = 0; i < 16; ++i) { a[i] = ap[i]; }
            accG[ms] = __builtin_amdgcn_wmma_f32_16x16x32_bf16(
                false, a, false, bG, (short)0, accG[ms], false, false);
            accU[ms] = __builtin_amdgcn_wmma_f32_16x16x32_bf16(
                false, a, false, bU, (short)0, accU[ms], false, false);
        }
        __syncthreads();
    }

    // Epilogue: h = silu(g) * u * topk_weight, store bf16 by compact slot.
#pragma unroll
    for (int ms = 0; ms < 4; ++ms) {
#pragma unroll
        for (int r = 0; r < 8; ++r) {
            int m16 = (lane < 16) ? r : (r + 8);
            int row = tileM + ms * 16 + m16;
            if (row < count) {
                float w = ewt[e * TOK + row];
                float g = accG[ms][r];
                float u = accU[ms][r];
                float s = g / (1.f + __expf(-g));
                hbuf[(size_t)(slotBase + row) * NF + f] = (__bf16)(s * u * w);
            }
        }
    }
}

// ---------------------------------------------------------------------------
// Down-projection: h[slots, F] @ down_w[e]^T -> atomicAdd into out[T, D].
__global__ __launch_bounds__(256) void moe_down(const __bf16* __restrict__ hbuf,
                                                const float* __restrict__ downW,
                                                const int* __restrict__ etok,
                                                const int* __restrict__ counts,
                                                float* __restrict__ out) {
    const int e     = blockIdx.z;
    const int count = counts[e];
    const int tileM = blockIdx.y * 64;
    if (tileM >= count) return;
    const int dBase    = blockIdx.x * 128;
    const int slotBase = prefix_count(counts, e);

    __shared__ __bf16 a_lds[64 * 32];

    const int tid  = threadIdx.x;
    const int wave = tid >> 5;
    const int lane = tid & 31;

    const int sm = tid >> 2;
    const int sq = tid & 3;
    const int srow   = tileM + sm;
    const bool svalid = srow < count;
    const __bf16* srcRow =
        hbuf + (size_t)(slotBase + (svalid ? srow : 0)) * NF + sq * 8;

    const int n     = lane & 15;
    const int khalf = (lane < 16) ? 0 : 16;
    const int kbase = (lane < 16) ? 0 : 16;
    const int arow  = (lane < 16) ? lane : (lane - 16);
    const int d     = dBase + wave * 16 + n;
    const float* wRow = downW + ((size_t)e * DIM + d) * NF + khalf;

    v8f acc[4];
    v8f vz = {};
#pragma unroll
    for (int i = 0; i < 4; ++i) acc[i] = vz;

    for (int k0 = 0; k0 < NF; k0 += 32) {
        uint4* dst = (uint4*)&a_lds[sm * 32 + sq * 8];
        if (svalid) *dst = *(const uint4*)(srcRow + k0);
        else        *dst = make_uint4(0u, 0u, 0u, 0u);
        __syncthreads();

        v16bf bW = load_cvt16(wRow + k0);

#pragma unroll
        for (int ms = 0; ms < 4; ++ms) {
            const __bf16* ap = &a_lds[(ms * 16 + arow) * 32 + kbase];
            v16bf a;
#pragma unroll
            for (int i = 0; i < 16; ++i) { a[i] = ap[i]; }
            acc[ms] = __builtin_amdgcn_wmma_f32_16x16x32_bf16(
                false, a, false, bW, (short)0, acc[ms], false, false);
        }
        __syncthreads();
    }

#pragma unroll
    for (int ms = 0; ms < 4; ++ms) {
#pragma unroll
        for (int r = 0; r < 8; ++r) {
            int m16 = (lane < 16) ? r : (r + 8);
            int row = tileM + ms * 16 + m16;
            if (row < count) {
                int tok = etok[e * TOK + row];
                atomicAdd(&out[(size_t)tok * DIM + d], acc[ms][r]);
            }
        }
    }
}

// ---------------------------------------------------------------------------
extern "C" void kernel_launch(void* const* d_in, const int* in_sizes, int n_in,
                              void* d_out, int out_size, void* d_ws, size_t ws_size,
                              hipStream_t stream) {
    (void)in_sizes; (void)n_in; (void)out_size; (void)ws_size;
    const float* x  = (const float*)d_in[0];  // [B,S,D]
    const float* rw = (const float*)d_in[1];  // [E,D]
    const float* gw = (const float*)d_in[2];  // [E,F,D]
    const float* uw = (const float*)d_in[3];  // [E,F,D]
    const float* dw = (const float*)d_in[4];  // [E,D,F]

    float* out        = (float*)d_out;                 // [T,D]
    float* logits_out = out + (size_t)TOK * DIM;       // [T,E]

    char*   ws     = (char*)d_ws;
    int*    counts = (int*)ws;                                   // 16 ints
    int*    etok   = (int*)(ws + 256);                           // E*T ints
    float*  ewt    = (float*)(ws + 256 + NEXP * TOK * 4);        // E*T floats
    __bf16* hbuf   = (__bf16*)(ws + 256 + NEXP * TOK * 8);       // T*2 x F bf16

    moe_zero<<<(TOK * DIM / 4 + 255) / 256, 256, 0, stream>>>(out, counts);
    moe_router<<<TOK / 16, 256, 0, stream>>>(x, rw, logits_out, counts, etok, ewt);

    dim3 gC(NF / 128, TOK / 64, NEXP);
    moe_gateup<<<gC, 256, 0, stream>>>(x, gw, uw, etok, ewt, counts, hbuf);

    dim3 gD(DIM / 128, TOK / 64, NEXP);
    moe_down<<<gD, 256, 0, stream>>>(hbuf, dw, etok, counts, out);
}
